// TrellisLinear_16999480558245
// MI455X (gfx1250) — matrix operand order repaired and checked by reading, not compiled, not run on hardware
//
#include <hip/hip_runtime.h>
#include <stdint.h>

typedef __attribute__((ext_vector_type(16))) _Float16 v16h;
typedef __attribute__((ext_vector_type(8)))  _Float16 v8h;
typedef __attribute__((ext_vector_type(4)))  _Float16 v4h;
typedef __attribute__((ext_vector_type(8)))  float    v8f;
typedef __attribute__((ext_vector_type(4)))  int      v4i;

// Address-space-qualified pointee types for CDNA5 async-copy builtins.
typedef __attribute__((address_space(1))) v4i as1_v4i;   // global
typedef __attribute__((address_space(3))) v4i as3_v4i;   // LDS

#define IN_DIM   4096
#define OUT_DIM  4096
#define MROWS    8192
#define ROWCHUNKS 32      // 8192 / 256

// ------------------------------------------------------------------
// CDNA5 async global->LDS helpers (guarded; fall back to sync copy)
// ------------------------------------------------------------------
__device__ __forceinline__ void copy16_to_lds(const _Float16* g, _Float16* l) {
#if defined(__AMDGCN__) && __has_builtin(__builtin_amdgcn_global_load_async_to_lds_b128)
  __builtin_amdgcn_global_load_async_to_lds_b128(
      (as1_v4i*)g, (as3_v4i*)l, 0, 0);
#else
  *(v8h*)l = *(const v8h*)g;
#endif
}

__device__ __forceinline__ void wait_async_lds() {
#if defined(__AMDGCN__) && __has_builtin(__builtin_amdgcn_global_load_async_to_lds_b128)
#if __has_builtin(__builtin_amdgcn_s_wait_asynccnt)
  __builtin_amdgcn_s_wait_asynccnt(0);
#else
  asm volatile("s_wait_asynccnt 0" ::: "memory");
#endif
#endif
}

// ------------------------------------------------------------------
// Kernel 1: partial per-column min/max (axis=0 over 8192 rows)
// ------------------------------------------------------------------
__global__ void k_colminmax(const float* __restrict__ x,
                            float* __restrict__ partmin,
                            float* __restrict__ partmax) {
  int c  = blockIdx.x * 256 + threadIdx.x;
  int r0 = blockIdx.y * 256;
  float mn = 3.4e38f, mx = -3.4e38f;
  const float* p = x + (size_t)r0 * IN_DIM + c;
  for (int r = 0; r < 256; ++r) {
    float v = p[(size_t)r * IN_DIM];
    mn = fminf(mn, v);
    mx = fmaxf(mx, v);
  }
  partmin[blockIdx.y * IN_DIM + c] = mn;
  partmax[blockIdx.y * IN_DIM + c] = mx;
}

// ------------------------------------------------------------------
// Kernel 2: finalize scale / zero-point per column
// ------------------------------------------------------------------
__global__ void k_scale(const float* __restrict__ partmin,
                        const float* __restrict__ partmax,
                        float* __restrict__ scale,
                        float* __restrict__ zp) {
  int c = blockIdx.x * 256 + threadIdx.x;
  float mn = 3.4e38f, mx = -3.4e38f;
  for (int k = 0; k < ROWCHUNKS; ++k) {
    mn = fminf(mn, partmin[k * IN_DIM + c]);
    mx = fmaxf(mx, partmax[k * IN_DIM + c]);
  }
  float s = (mx - mn) * (1.0f / 255.0f);
  if (!(s > 0.0f)) s = 1.0f;
  scale[c] = s;
  zp[c]    = rintf(-mn / s);
}

// ------------------------------------------------------------------
// Kernel 3: fake-quant x -> f16 (4 elements / thread, vectorized)
// ------------------------------------------------------------------
__global__ void k_quant(const float* __restrict__ x,
                        const float* __restrict__ scale,
                        const float* __restrict__ zp,
                        _Float16* __restrict__ xq) {
  size_t i4 = ((size_t)blockIdx.x * 256 + threadIdx.x) * 4;
  int c = (int)(i4 & (IN_DIM - 1));
  float4 v = *(const float4*)(x + i4);
  v4h o;
  float vv[4] = {v.x, v.y, v.z, v.w};
  #pragma unroll
  for (int j = 0; j < 4; ++j) {
    float s = scale[c + j];
    float z = zp[c + j];
    float q = fminf(fmaxf(rintf(vv[j] / s) + z, 0.0f), 255.0f);
    o[j] = (_Float16)((q - z) * s);
  }
  *(v4h*)(xq + i4) = o;
}

// ------------------------------------------------------------------
// Kernel 4: trellis decode + sv/su signs + row FWHT (over IN) -> f16
// ------------------------------------------------------------------
__global__ void k_decode_rows(const int* __restrict__ trellis,
                              const float* __restrict__ lut,
                              const float* __restrict__ su,
                              const float* __restrict__ sv,
                              _Float16* __restrict__ wh) {
  __shared__ float row[IN_DIM];
  const int o   = blockIdx.x;
  const int tid = threadIdx.x;
  const float svo = sv[o];
  const int tbase = o * 16;           // first trellis row of this W row
  for (int s = tid; s < 2048; s += 256) {
    int tr = s >> 7;                  // trellis row within this W row
    int i  = s & 127;                 // state index within trellis row
    const int* u = trellis + (size_t)(tbase + tr) * 32;
    int word = i >> 2;
    int off  = (i & 3) * 4;           // start bit = i*4, off = start%16
    uint32_t hi = ((uint32_t)u[word]) & 0xFFFFu;
    uint32_t lo = ((uint32_t)u[(word + 1) & 31]) & 0xFFFFu;  // cyclic wrap
    uint32_t st = ((hi << off) | (lo >> (16 - off))) & 0xFFFFu;
    float v0 = lut[st * 2 + 0];
    float v1 = lut[st * 2 + 1];
    int pos = tr * 256 + i * 2;
    row[pos]     = svo * su[pos]     * v0;
    row[pos + 1] = svo * su[pos + 1] * v1;
  }
  __syncthreads();
  for (int h = 1; h < IN_DIM; h <<= 1) {
    for (int q = tid; q < IN_DIM / 2; q += 256) {
      int i = ((q & ~(h - 1)) << 1) | (q & (h - 1));
      float a = row[i], b = row[i + h];
      row[i]     = a + b;
      row[i + h] = a - b;
    }
    __syncthreads();
  }
  for (int i = tid; i < IN_DIM; i += 256)
    wh[(size_t)o * IN_DIM + i] = (_Float16)(row[i] * 0.015625f);  // 1/64
}

// ------------------------------------------------------------------
// Kernel 5: column FWHT (over OUT) + per-group scale, in place on Wh.
// ------------------------------------------------------------------
__global__ void k_fwht_cols(_Float16* __restrict__ wh,
                            const float* __restrict__ wscales) {
  extern __shared__ float tile[];     // OUT_DIM * 8 floats (128 KB)
  const int c0  = blockIdx.x * 8;
  const int tid = threadIdx.x;
  for (int idx = tid; idx < OUT_DIM * 8; idx += 256) {
    int r = idx >> 3, j = idx & 7;
    tile[idx] = (float)wh[(size_t)r * IN_DIM + c0 + j];
  }
  __syncthreads();
  for (int h = 1; h < OUT_DIM; h <<= 1) {
    for (int idx = tid; idx < (OUT_DIM / 2) * 8; idx += 256) {
      int q = idx >> 3, j = idx & 7;
      int i = ((q & ~(h - 1)) << 1) | (q & (h - 1));
      float a = tile[i * 8 + j], b = tile[(i + h) * 8 + j];
      tile[i * 8 + j]       = a + b;
      tile[(i + h) * 8 + j] = a - b;
    }
    __syncthreads();
  }
  const int sgrp = c0 >> 8;
  for (int idx = tid; idx < OUT_DIM * 8; idx += 256) {
    int r = idx >> 3, j = idx & 7;
    float sc = wscales[r * 16 + sgrp];
    wh[(size_t)r * IN_DIM + c0 + j] = (_Float16)(tile[idx] * 0.015625f * sc);
  }
}

// ------------------------------------------------------------------
// Kernel 6: WMMA GEMM  C(8192x4096) = Xq @ Wh^T
// Block tile 256x128, BK=32, 8 wave32s in 4x2; 64x64 per wave.
// #pragma unroll 1 on the K loop keeps VGPR allocation < 256
// (acc 128 + af 32 + bf 16 live at a time).
// Double-buffered LDS fed by GLOBAL_LOAD_ASYNC_TO_LDS_B128.
// ------------------------------------------------------------------
#define BM 256
#define BN 128
#define BK 32
#define LDA 40                 // halves; 80B row stride -> conflict-free frags
#define ASZ (BM * LDA)         // 10240 halves / buffer
#define BSZ (BN * LDA)         // 5120 halves / buffer

__global__ __launch_bounds__(256) void k_gemm(const _Float16* __restrict__ A,
                                              const _Float16* __restrict__ B,
                                              float* __restrict__ C) {
  __shared__ __align__(16) _Float16 As[2 * ASZ];   // 40 KB
  __shared__ __align__(16) _Float16 Bs[2 * BSZ];   // 20 KB
  const int tid  = threadIdx.x;
  const int lane = tid & 31;
  const int wave = tid >> 5;
  const int lm = lane & 15;           // M (A) / N (B) index within 16x16 tile
  const int lk = lane >> 4;           // K half selector
  const int wm0 = (wave >> 1) * 64;   // wave row origin in block tile
  const int wn0 = (wave & 1) * 64;    // wave col origin in block tile
  const int m0 = blockIdx.y * BM;
  const int n0 = blockIdx.x * BN;
  const int K  = IN_DIM;

  v8f acc[4][4];
  #pragma unroll
  for (int a = 0; a < 4; ++a)
    #pragma unroll
    for (int b = 0; b < 4; ++b) { v8f z = {}; acc[a][b] = z; }

  // stage one 256x32 A tile + 128x32 B tile into LDS buffer `buf`
  auto stage = [&](int buf, int k0) {
    _Float16* as = As + buf * ASZ;
    _Float16* bs = Bs + buf * BSZ;
    #pragma unroll
    for (int l = 0; l < 4; ++l) {                  // 1024 16B chunks of A
      int cidx = tid + l * 256;
      int r = cidx >> 2, seg = cidx & 3;
      copy16_to_lds(A + (size_t)(m0 + r) * K + k0 + seg * 8,
                    as + r * LDA + seg * 8);
    }
    #pragma unroll
    for (int l = 0; l < 2; ++l) {                  // 512 16B chunks of B
      int cidx = tid + l * 256;
      int r = cidx >> 2, seg = cidx & 3;
      copy16_to_lds(B + (size_t)(n0 + r) * K + k0 + seg * 8,
                    bs + r * LDA + seg * 8);
    }
  };

  stage(0, 0);
  wait_async_lds();
  __syncthreads();

  union Frag { v16h v; v8h h[2]; };
  const int KT = K / BK;
  #pragma unroll 1               // no unroll: keep VGPR live ranges short
  for (int kt = 0; kt < KT; ++kt) {
    const int buf = kt & 1;
    if (kt + 1 < KT) stage(buf ^ 1, (kt + 1) * BK);  // overlap with compute

    const _Float16* as = As + buf * ASZ;
    const _Float16* bs = Bs + buf * BSZ;

    Frag af[4];
    #pragma unroll
    for (int t = 0; t < 4; ++t) {
      const _Float16* p = as + (wm0 + t * 16 + lm) * LDA;
      af[t].h[0] = *(const v8h*)(p + lk * 8);        // K = kh..kh+7
      af[t].h[1] = *(const v8h*)(p + 16 + lk * 8);   // K = 16+kh..16+kh+7
    }
    // process N in two halves so only 2 B fragments are live at a time
    #pragma unroll
    for (int half = 0; half < 2; ++half) {
      Frag bf[2];
      #pragma unroll
      for (int t = 0; t < 2; ++t) {
        const _Float16* q = bs + (wn0 + (half * 2 + t) * 16 + lm) * LDA;
        bf[t].h[0] = *(const v8h*)(q + lk * 8);        // K = kh..kh+7
        bf[t].h[1] = *(const v8h*)(q + 16 + lk * 8);   // K = 16+kh..16+kh+7
      }
      #pragma unroll
      for (int tm = 0; tm < 4; ++tm)
        #pragma unroll
        for (int tn = 0; tn < 2; ++tn)
          acc[tm][half * 2 + tn] = __builtin_amdgcn_wmma_f32_16x16x32_f16(
              false, af[tm].v, false, bf[tn].v, (short)0,
              acc[tm][half * 2 + tn], false, false);
    }

    if (kt + 1 < KT) {
      wait_async_lds();   // my async writes to buf^1 are complete
      __syncthreads();    // everyone's writes done; everyone done reading buf
    }
  }

  // epilogue: C/D layout: lane -> N = lm, VGPR r -> M = 8*lk + r
  #pragma unroll
  for (int tm = 0; tm < 4; ++tm)
    #pragma unroll
    for (int tn = 0; tn < 4; ++tn)
      #pragma unroll
      for (int r = 0; r < 8; ++r) {
        int m = m0 + wm0 + tm * 16 + lk * 8 + r;
        int n = n0 + wn0 + tn * 16 + lm;
        C[(size_t)m * OUT_DIM + n] = acc[tm][tn][r];
      }
}

// ------------------------------------------------------------------
// Launch
// ------------------------------------------------------------------
extern "C" void kernel_launch(void* const* d_in, const int* in_sizes, int n_in,
                              void* d_out, int out_size, void* d_ws, size_t ws_size,
                              hipStream_t stream) {
  (void)in_sizes; (void)n_in; (void)out_size; (void)ws_size;
  const float* x       = (const float*)d_in[0];
  const int*   trellis = (const int*)d_in[1];
  const float* su      = (const float*)d_in[2];
  const float* sv      = (const float*)d_in[3];
  const float* wscales = (const float*)d_in[4];
  const float* lut     = (const float*)d_in[5];
  float* out = (float*)d_out;

  char* ws = (char*)d_ws;
  size_t off = 0;
  auto take = [&](size_t bytes) -> char* {
    size_t o = (off + 255) & ~(size_t)255;
    off = o + bytes;
    return ws + o;
  };
  float*    partmin = (float*)take((size_t)ROWCHUNKS * IN_DIM * 4);
  float*    partmax = (float*)take((size_t)ROWCHUNKS * IN_DIM * 4);
  float*    scale   = (float*)take((size_t)IN_DIM * 4);
  float*    zp      = (float*)take((size_t)IN_DIM * 4);
  _Float16* xq      = (_Float16*)take((size_t)MROWS * IN_DIM * 2);    // 64 MB
  _Float16* wh      = (_Float16*)take((size_t)OUT_DIM * IN_DIM * 2);  // 32 MB

  k_colminmax<<<dim3(IN_DIM / 256, ROWCHUNKS), 256, 0, stream>>>(x, partmin, partmax);
  k_scale<<<IN_DIM / 256, 256, 0, stream>>>(partmin, partmax, scale, zp);
  k_quant<<<(int)(((size_t)MROWS * IN_DIM) / (256 * 4)), 256, 0, stream>>>(x, scale, zp, xq);
  k_decode_rows<<<OUT_DIM, 256, 0, stream>>>(trellis, lut, su, sv, wh);
  k_fwht_cols<<<IN_DIM / 8, 256, OUT_DIM * 8 * sizeof(float), stream>>>(wh, wscales);
  k_gemm<<<dim3(OUT_DIM / BN, MROWS / BM), 256, 0, stream>>>(xq, wh, out);
}
